// Equiformer_MD17_DeNS_84602265797130
// MI455X (gfx1250) — matrix operand order repaired and checked by reading, not compiled, not run on hardware
//
#include <hip/hip_runtime.h>
#include <math.h>

typedef __attribute__((ext_vector_type(16))) _Float16 v16h;
typedef __attribute__((ext_vector_type(8)))  _Float16 v8h;
typedef __attribute__((ext_vector_type(4)))  _Float16 v4h;
typedef __attribute__((ext_vector_type(8)))  float    v8f;

#define CUTOFF_R 5.0f
#define NRBF 32

__device__ __forceinline__ float siluf(float x) { return x / (1.f + __expf(-x)); }

// ---------------------------------------------------------------------------
// Edge geometry: unit vector, distance, real spherical harmonics l=0..2, RBF
// ---------------------------------------------------------------------------
__global__ void edge_geom_kernel(const float* __restrict__ pos,
                                 const int* __restrict__ src,
                                 const int* __restrict__ dst,
                                 float* __restrict__ sh,
                                 float* __restrict__ rbf,
                                 float* __restrict__ dist,
                                 float* __restrict__ uvec,
                                 int E) {
  int e = blockIdx.x * blockDim.x + threadIdx.x;
  if (e >= E) return;
  int s = src[e], d = dst[e];
  float vx = pos[s * 3 + 0] - pos[d * 3 + 0];
  float vy = pos[s * 3 + 1] - pos[d * 3 + 1];
  float vz = pos[s * 3 + 2] - pos[d * 3 + 2];
  float r = sqrtf(vx * vx + vy * vy + vz * vz + 1e-12f);
  float ir = 1.f / r;
  float x = vx * ir, y = vy * ir, z = vz * ir;
  dist[e] = r;
  uvec[e * 3 + 0] = x; uvec[e * 3 + 1] = y; uvec[e * 3 + 2] = z;
  float* S = sh + (size_t)e * 9;
  const float c3 = 1.7320508f, c15 = 3.8729833f, c5h = 1.1180340f;
  S[0] = 1.f;
  S[1] = c3 * y; S[2] = c3 * z; S[3] = c3 * x;
  S[4] = c15 * x * y; S[5] = c15 * y * z; S[6] = c5h * (3.f * z * z - 1.f);
  S[7] = c15 * x * z; S[8] = 0.5f * c15 * (x * x - y * y);
  float start = __expf(-CUTOFF_R);
  float tw = (2.f / NRBF) * (1.f - start);
  float beta = 1.f / (tw * tw);
  float cut = (r < CUTOFF_R) ? 0.5f * (__cosf(3.14159265358979f * r / CUTOFF_R) + 1.f) : 0.f;
  float er = __expf(-r);
  float* R = rbf + (size_t)e * NRBF;
  for (int i = 0; i < NRBF; ++i) {
    float mean = start + (1.f - start) * i / (NRBF - 1);
    float df = er - mean;
    R[i] = cut * __expf(-beta * df * df);
  }
}

// ---------------------------------------------------------------------------
// Generic strided WMMA GEMM: C[M,N] (+)= act(A[M,K] * B[K,N])
// 256 threads = 8 wave32 waves in a 4x2 grid; block tile 64x64, wave tile
// 16x32 (two v_wmma_f32_16x16x32_f16 per K-step sharing the A fragment).
// A staged [row][k] f16; B staged TRANSPOSED [col][k] f16 so every fragment
// is contiguous in LDS: A = 2x ds_load_b128, B = 2x ds_load_b128 per tile.
// act: 0=none 1=silu 2=leaky(0.2) 3=sigmoid ; accm: 0=store 1=atomicAdd
// ---------------------------------------------------------------------------
__global__ void wmma_gemm_kernel(const float* __restrict__ A,
                                 const float* __restrict__ B,
                                 float* __restrict__ C,
                                 int M, int N, int K,
                                 int lar, int lac, int lbr, int lbc,
                                 int lcr, int lcc, int act, int accm) {
  __shared__ __align__(16) _Float16 Ash[64][40];   // 64 rows x 32 k (+pad)
  __shared__ __align__(16) _Float16 BshT[64][40];  // 64 cols x 32 k (+pad)
  int tid = threadIdx.x;
  int wave = tid >> 5, lane = tid & 31;
  int mw = wave >> 1, nw = wave & 1;   // 4x2 wave grid
  int g = lane >> 4, hl = lane & 15;   // lane half-group, lane-in-half
  int m0 = blockIdx.y * 64, n0 = blockIdx.x * 64;
  v8f acc0 = {}, acc1 = {};
  for (int k0 = 0; k0 < K; k0 += 32) {
    if (k0 + 32 < K) {  // prefetch next A tile -> global_prefetch_b8
      int gm = m0 + (tid >> 2), gk = k0 + 32 + ((tid & 3) << 3);
      if (gm < M && gk < K) __builtin_prefetch(&A[(size_t)gm * lar + (size_t)gk * lac], 0, 1);
    }
    // ---- stage A (64x32) ----
    if (lac == 1) {
      for (int i = tid; i < 64 * 8; i += 256) {  // float4 chunks
        int m = i >> 3, kc = (i & 7) << 2;
        int gm = m0 + m, gk = k0 + kc;
        v4h h = {};
        if (gm < M && gk + 3 < K) {
          float4 f = *(const float4*)&A[(size_t)gm * lar + gk];
          h[0] = (_Float16)f.x; h[1] = (_Float16)f.y;
          h[2] = (_Float16)f.z; h[3] = (_Float16)f.w;
        } else if (gm < M) {
          for (int j = 0; j < 4; ++j)
            if (gk + j < K) h[j] = (_Float16)A[(size_t)gm * lar + gk + j];
        }
        *(v4h*)&Ash[m][kc] = h;
      }
    } else {
      for (int i = tid; i < 64 * 32; i += 256) {
        int m = i >> 5, k = i & 31;
        int gm = m0 + m, gk = k0 + k;
        float v = (gm < M && gk < K) ? A[(size_t)gm * lar + (size_t)gk * lac] : 0.f;
        Ash[m][k] = (_Float16)v;
      }
    }
    // ---- stage B transposed (cols x k) ----
    if (lbc == 1) {
      for (int i = tid; i < 32 * 16; i += 256) {  // k x (64/4) float4 chunks
        int k = i >> 4, nc = (i & 15) << 2;
        int gk = k0 + k, gn = n0 + nc;
        if (gk < K && gn + 3 < N) {
          float4 f = *(const float4*)&B[(size_t)gk * lbr + gn];
          BshT[nc + 0][k] = (_Float16)f.x;
          BshT[nc + 1][k] = (_Float16)f.y;
          BshT[nc + 2][k] = (_Float16)f.z;
          BshT[nc + 3][k] = (_Float16)f.w;
        } else {
          for (int j = 0; j < 4; ++j) {
            float v = (gk < K && gn + j < N) ? B[(size_t)gk * lbr + gn + j] : 0.f;
            BshT[nc + j][k] = (_Float16)v;
          }
        }
      }
    } else {
      for (int i = tid; i < 64 * 32; i += 256) {
        int n = i >> 5, k = i & 31;
        int gk = k0 + k, gn = n0 + n;
        float v = (gk < K && gn < N) ? B[(size_t)gk * lbr + (size_t)gn * lbc] : 0.f;
        BshT[n][k] = (_Float16)v;
      }
    }
    __syncthreads();
    // ---- fragments: contiguous LDS reads (ds_load_b128) ----
    const v8h* arow = (const v8h*)&Ash[(mw << 4) + hl][0];   // 5 v8h per row
    v8h alo = arow[g << 1];       // K = g*16 + 0..7
    v8h ahi = arow[(g << 1) + 1]; // K = g*16 + 8..15
    v16h a = __builtin_shufflevector(alo, ahi, 0, 1, 2, 3, 4, 5, 6, 7,
                                     8, 9, 10, 11, 12, 13, 14, 15);
    const v8h* bcol0 = (const v8h*)&BshT[(nw << 5) + hl][0];
    const v8h* bcol1 = (const v8h*)&BshT[(nw << 5) + 16 + hl][0];
    v16h b0 = __builtin_shufflevector(bcol0[g << 1], bcol0[(g << 1) + 1],
                                      0, 1, 2, 3, 4, 5, 6, 7, 8, 9, 10, 11, 12, 13, 14, 15);
    v16h b1 = __builtin_shufflevector(bcol1[g << 1], bcol1[(g << 1) + 1],
                                      0, 1, 2, 3, 4, 5, 6, 7, 8, 9, 10, 11, 12, 13, 14, 15);
    acc0 = __builtin_amdgcn_wmma_f32_16x16x32_f16(false, a, false, b0, (short)0, acc0, false, false);
    acc1 = __builtin_amdgcn_wmma_f32_16x16x32_f16(false, a, false, b1, (short)0, acc1, false, false);
    __syncthreads();
  }
#pragma unroll
  for (int r = 0; r < 8; ++r) {
    int gm = m0 + (mw << 4) + r + (g << 3);
    if (gm >= M) continue;
#pragma unroll
    for (int half = 0; half < 2; ++half) {
      int gn = n0 + (nw << 5) + (half << 4) + hl;
      if (gn >= N) continue;
      float v = half ? acc1[r] : acc0[r];
      if (act == 1) v = siluf(v);
      else if (act == 2) v = v > 0.f ? v : 0.2f * v;
      else if (act == 3) v = 1.f / (1.f + __expf(-v));
      float* p = &C[(size_t)gm * lcr + (size_t)gn * lcc];
      if (accm) atomicAdd(p, v); else *p = v;
    }
  }
}

__global__ void zero_kernel(float* p, long n) {
  long i = (long)blockIdx.x * blockDim.x + threadIdx.x;
  if (i < n) p[i] = 0.f;
}

__global__ void embed_kernel(const float* __restrict__ emb, const int* __restrict__ z,
                             float* __restrict__ s, int N) {
  int i = blockIdx.x * blockDim.x + threadIdx.x;
  if (i >= N * 128) return;
  int n = i >> 7, c = i & 127;
  s[i] = emb[z[n] * 128 + c];
}

// degree-embedding scatter: m0/m1/m2 = w * sh, segment-summed into node temps
__global__ void deg_scatter_kernel(const float* __restrict__ rw, const float* __restrict__ sh,
                                   const int* __restrict__ dst,
                                   float* t0, float* t1, float* t2, int E, float inv) {
  int e = blockIdx.x;
  int u = threadIdx.x;
  if (e >= E) return;
  const float* W = rw + (size_t)e * 960;   // first 224 cols used
  const float* S = sh + (size_t)e * 9;
  int d = dst[e];
  if (u < 128) {
    atomicAdd(&t0[(size_t)d * 128 + u], W[u] * S[0] * inv);
  } else if (u < 320) {
    int j = u - 128; int c = j / 3, m = j % 3;
    atomicAdd(&t1[((size_t)d * 64 + c) * 3 + m], W[128 + c] * S[1 + m] * inv);
  } else if (u < 480) {
    int j = u - 320; int c = j / 5, m = j % 5;
    atomicAdd(&t2[((size_t)d * 32 + c) * 5 + m], W[192 + c] * S[4 + m] * inv);
  }
}

__global__ void gather_msgs_kernel(const float* fs0, const float* fd0,
                                   const float* fs1, const float* fd1,
                                   const float* fs2, const float* fd2,
                                   const int* __restrict__ src, const int* __restrict__ dst,
                                   float* msg0, float* msg1, float* msg2, int E) {
  int e = blockIdx.x;
  int u = threadIdx.x;
  if (e >= E) return;
  int s = src[e], d = dst[e];
  if (u < 128) msg0[(size_t)e * 128 + u] = fs0[(size_t)s * 128 + u] + fd0[(size_t)d * 128 + u];
  else if (u < 320) { int j = u - 128; msg1[(size_t)e * 192 + j] = fs1[(size_t)s * 192 + j] + fd1[(size_t)d * 192 + j]; }
  else if (u < 480) { int j = u - 320; msg2[(size_t)e * 160 + j] = fs2[(size_t)s * 160 + j] + fd2[(size_t)d * 160 + j]; }
}

// ---------------------------------------------------------------------------
// Depthwise tensor product over the 15 CG paths (l1,l2,l3), weighted by w[e,960]
// ---------------------------------------------------------------------------
__global__ void dtp_kernel(const float* __restrict__ msg0, const float* __restrict__ msg1,
                           const float* __restrict__ msg2, const float* __restrict__ sh,
                           const float* __restrict__ w,
                           float* __restrict__ d0, float* __restrict__ d1,
                           float* __restrict__ d2, int E) {
  int e = blockIdx.x;
  int u = threadIdx.x;  // 0..127
  if (e >= E) return;
  const float* S = sh + (size_t)e * 9;
  const float* W = w + (size_t)e * 960;
  const float* M0 = msg0 + (size_t)e * 128;
  const float* M1 = msg1 + (size_t)e * 192;
  const float* M2 = msg2 + (size_t)e * 160;
  float* D0 = d0 + (size_t)e * 224;
  float* D1 = d1 + (size_t)e * 1152;
  float* D2 = d2 + (size_t)e * 1760;
  float s0 = S[0];
  const float* s1 = S + 1;
  const float* s2 = S + 4;

  float m = M0[u];
  D0[u] = m * s0 * W[u];                                   // (0,0,0)
  { float wp = W[128 + u];                                 // (0,1,1)
    for (int k = 0; k < 3; ++k) D1[u * 3 + k] = m * s1[k] * wp; }
  { float wp = W[256 + u];                                 // (0,2,2)
    for (int k = 0; k < 5; ++k) D2[u * 5 + k] = m * s2[k] * wp; }

  if (u < 64) {
    const float* a = M1 + u * 3;
    { float wp = W[384 + u];                               // (1,0,1)
      for (int k = 0; k < 3; ++k) D1[(128 + u) * 3 + k] = a[k] * s0 * wp; }
    { float wp = W[448 + u];                               // (1,1,0)
      float dt = (a[0] * s1[0] + a[1] * s1[1] + a[2] * s1[2]) * 0.57735027f;
      D0[128 + u] = dt * wp; }
    { float wp = W[512 + u] * 0.70710678f;                 // (1,1,1) cross
      D1[(192 + u) * 3 + 0] = (a[1] * s1[2] - a[2] * s1[1]) * wp;
      D1[(192 + u) * 3 + 1] = (a[2] * s1[0] - a[0] * s1[2]) * wp;
      D1[(192 + u) * 3 + 2] = (a[0] * s1[1] - a[1] * s1[0]) * wp; }
    { float wp = W[576 + u] * 0.5f;                        // (1,1,2)
      for (int k = 0; k < 5; ++k) D2[(128 + u) * 5 + k] = a[k % 3] * s1[(k + 1) % 3] * wp; }
    { float wp = W[640 + u] * 0.5f;                        // (1,2,1)
      for (int k = 0; k < 3; ++k) D1[(256 + u) * 3 + k] = a[(k + 1) % 3] * s2[k] * wp; }
    { float wp = W[704 + u] * 0.5f;                        // (1,2,2)
      for (int k = 0; k < 5; ++k) D2[(192 + u) * 5 + k] = a[k % 3] * s2[(k + 2) % 5] * wp; }
  }
  if (u < 32) {
    const float* b = M2 + u * 5;
    { float wp = W[768 + u];                               // (2,0,2)
      for (int k = 0; k < 5; ++k) D2[(256 + u) * 5 + k] = b[k] * s0 * wp; }
    { float wp = W[800 + u] * 0.5f;                        // (2,1,1)
      for (int k = 0; k < 3; ++k) D1[(320 + u) * 3 + k] = b[(k + 1) % 5] * s1[k] * wp; }
    { float wp = W[832 + u] * 0.5f;                        // (2,1,2)
      for (int k = 0; k < 5; ++k) D2[(288 + u) * 5 + k] = b[k] * s1[k % 3] * wp; }
    { float wp = W[864 + u];                               // (2,2,0)
      float dt = 0.f;
      for (int k = 0; k < 5; ++k) dt += b[k] * s2[k];
      D0[192 + u] = dt * 0.4472136f * wp; }
    { float wp = W[896 + u] * 0.5f;                        // (2,2,1)
      for (int k = 0; k < 3; ++k) D1[(352 + u) * 3 + k] = b[k] * s2[(k + 1) % 5] * wp; }
    { float wp = W[928 + u] * 0.5f;                        // (2,2,2)
      for (int k = 0; k < 5; ++k) D2[(320 + u) * 5 + k] = b[(k + 2) % 5] * s2[(k + 3) % 5] * wp; }
  }
}

// ---------------------------------------------------------------------------
// Attention: per-(edge,head) logit, segment max/sum via atomics, alpha, scatter
// ---------------------------------------------------------------------------
__global__ void logits_kernel(const float* __restrict__ h0, const float* __restrict__ a,
                              float* __restrict__ logits, unsigned* __restrict__ mxenc,
                              const int* __restrict__ dst, int E) {
  int i = blockIdx.x * blockDim.x + threadIdx.x;
  if (i >= E * 4) return;
  int e = i >> 2, h = i & 3;
  const float* H = h0 + (size_t)e * 128 + h * 32;
  const float* A = a + h * 32;
  float acc = 0.f;
  for (int c = 0; c < 32; ++c) {
    float x = H[c];
    x = x > 0.f ? x : 0.2f * x;
    acc += x * A[c];
  }
  logits[i] = acc;
  unsigned bits = __float_as_uint(acc);
  unsigned enc = (bits & 0x80000000u) ? ~bits : (bits | 0x80000000u);
  atomicMax(&mxenc[(size_t)dst[e] * 4 + h], enc);
}

__global__ void expsum_kernel(const float* __restrict__ logits, const unsigned* __restrict__ mxenc,
                              const int* __restrict__ dst, float* __restrict__ ex,
                              float* __restrict__ den, int E) {
  int i = blockIdx.x * blockDim.x + threadIdx.x;
  if (i >= E * 4) return;
  int e = i >> 2, h = i & 3;
  unsigned enc = mxenc[(size_t)dst[e] * 4 + h];
  float mx = (enc & 0x80000000u) ? __uint_as_float(enc & 0x7fffffffu) : __uint_as_float(~enc);
  float v = __expf(logits[i] - mx);
  ex[i] = v;
  atomicAdd(&den[(size_t)dst[e] * 4 + h], v);
}

__global__ void alpha_kernel(const float* __restrict__ ex, const float* __restrict__ den,
                             const int* __restrict__ dst, float* __restrict__ alp, int E) {
  int i = blockIdx.x * blockDim.x + threadIdx.x;
  if (i >= E * 4) return;
  int e = i >> 2, h = i & 3;
  alp[i] = ex[i] / (den[(size_t)dst[e] * 4 + h] + 1e-9f);
}

__global__ void attn_scatter_kernel(const float* __restrict__ h0, const float* __restrict__ h1,
                                    const float* __restrict__ h2, const float* __restrict__ alpha,
                                    const int* __restrict__ dst,
                                    float* o0, float* o1, float* o2, int E) {
  int e = blockIdx.x;
  int u = threadIdx.x;
  if (e >= E) return;
  int d = dst[e];
  const float* al = alpha + (size_t)e * 4;
  if (u < 128) {
    atomicAdd(&o0[(size_t)d * 128 + u], al[u >> 5] * h0[(size_t)e * 128 + u]);
  } else if (u < 320) {
    int j = u - 128; int c = j / 3;
    atomicAdd(&o1[(size_t)d * 192 + j], al[c >> 4] * h1[(size_t)e * 192 + j]);
  } else if (u < 480) {
    int j = u - 320; int c = j / 5;
    atomicAdd(&o2[(size_t)d * 160 + j], al[c >> 3] * h2[(size_t)e * 160 + j]);
  }
}

// ---------------------------------------------------------------------------
// Equivariant norms
// ---------------------------------------------------------------------------
__global__ void eqnorm_s_kernel(const float* __restrict__ s, float* __restrict__ out,
                                const float* __restrict__ gain, int N) {
  __shared__ float red[128];
  int n = blockIdx.x, u = threadIdx.x;
  if (n >= N) return;
  float x = s[(size_t)n * 128 + u];
  red[u] = x; __syncthreads();
  for (int st = 64; st > 0; st >>= 1) { if (u < st) red[u] += red[u + st]; __syncthreads(); }
  float mean = red[0] / 128.f;
  __syncthreads();
  float x0 = x - mean;
  red[u] = x0 * x0; __syncthreads();
  for (int st = 64; st > 0; st >>= 1) { if (u < st) red[u] += red[u + st]; __syncthreads(); }
  float inv = rsqrtf(red[0] / 128.f + 1e-5f);
  out[(size_t)n * 128 + u] = x0 * inv * gain[u];
}

__global__ void eqnorm_vt_kernel(const float* __restrict__ x, float* __restrict__ out,
                                 const float* __restrict__ gain, int N, int C, int Mdim) {
  __shared__ float red[256];
  int n = blockIdx.x, t = threadIdx.x;
  if (n >= N) return;
  int CM = C * Mdim;
  float acc = 0.f;
  for (int i = t; i < CM; i += 256) { float v = x[(size_t)n * CM + i]; acc += v * v; }
  red[t] = acc; __syncthreads();
  for (int st = 128; st > 0; st >>= 1) { if (t < st) red[t] += red[t + st]; __syncthreads(); }
  float inv = rsqrtf(red[0] / C + 1e-5f);
  for (int i = t; i < CM; i += 256) {
    int c = i / Mdim;
    out[(size_t)n * CM + i] = x[(size_t)n * CM + i] * inv * gain[c];
  }
}

__global__ void gate_apply_kernel(float* v1, float* t1, const float* __restrict__ gate, int N) {
  int n = blockIdx.x, u = threadIdx.x;
  if (n >= N) return;
  const float* g = gate + (size_t)n * 96;
  if (u < 192) { int c = u / 3; v1[(size_t)n * 192 + u] *= g[c]; }
  else if (u < 352) { int j = u - 192; int c = j / 5; t1[(size_t)n * 160 + j] *= g[64 + c]; }
}

__global__ void node_energy_kernel(const float* __restrict__ hid, const float* __restrict__ e2,
                                   const int* __restrict__ batch, float* outE, int N, float scale) {
  int n = blockIdx.x * blockDim.x + threadIdx.x;
  if (n >= N) return;
  float acc = 0.f;
  for (int c = 0; c < 128; ++c) acc += hid[(size_t)n * 128 + c] * e2[c];
  atomicAdd(&outE[batch[n]], acc * scale);
}

// backward force pass: differentiate cutoff*RBF chain per edge, scatter +-f
__global__ void force_kernel(const float* __restrict__ dist, const float* __restrict__ uvec,
                             const float* __restrict__ rw,
                             const int* __restrict__ src, const int* __restrict__ dst,
                             float* force, int E) {
  int e = blockIdx.x * blockDim.x + threadIdx.x;
  if (e >= E) return;
  float r = dist[e];
  float start = __expf(-CUTOFF_R);
  float tw = (2.f / NRBF) * (1.f - start);
  float beta = 1.f / (tw * tw);
  float pi = 3.14159265358979f;
  float cut = (r < CUTOFF_R) ? 0.5f * (__cosf(pi * r / CUTOFF_R) + 1.f) : 0.f;
  float dcut = (r < CUTOFF_R) ? -0.5f * (pi / CUTOFF_R) * __sinf(pi * r / CUTOFF_R) : 0.f;
  float er = __expf(-r);
  const float* W = rw + (size_t)e * 960;
  float g = 0.f;
  for (int i = 0; i < NRBF; ++i) {
    float mean = start + (1.f - start) * i / (NRBF - 1);
    float df = er - mean;
    float gaus = __expf(-beta * df * df);
    float drbf = dcut * gaus + cut * gaus * (2.f * beta * df) * er;
    g += drbf * W[i * 30];
  }
  int s = src[e], d = dst[e];
  for (int k = 0; k < 3; ++k) {
    float f = -g * uvec[e * 3 + k];
    atomicAdd(&force[s * 3 + k], f);
    atomicAdd(&force[d * 3 + k], -f);
  }
}

// ---------------------------------------------------------------------------
static inline void gemm(hipStream_t st, const float* A, const float* B, float* C,
                        int M, int Nn, int K, int lar, int lac, int lbr, int lbc,
                        int lcr, int lcc, int act, int accm) {
  dim3 g((Nn + 63) / 64, (M + 63) / 64);
  wmma_gemm_kernel<<<g, 256, 0, st>>>(A, B, C, M, Nn, K, lar, lac, lbr, lbc, lcr, lcc, act, accm);
}

static inline void zero(hipStream_t st, float* p, long n) {
  zero_kernel<<<(unsigned)((n + 255) / 256), 256, 0, st>>>(p, n);
}

extern "C" void kernel_launch(void* const* d_in, const int* in_sizes, int n_in,
                              void* d_out, int out_size, void* d_ws, size_t ws_size,
                              hipStream_t stream) {
  const float* pos = (const float*)d_in[0];
  int N = in_sizes[0] / 3;
  const int* z     = (const int*)d_in[n_in - 4];
  const int* batch = (const int*)d_in[n_in - 3];
  const int* src   = (const int*)d_in[n_in - 2];
  const int* dst   = (const int*)d_in[n_in - 1];
  int E = in_sizes[n_in - 2];

  // pytree(alphabetical) flatten: [1]=atom_emb, [2..30]=blocks leaves,
  // [31..36]=deg{p0,p1,p2,r1,r2,r3}, [37]=fns, [40..42]=head{b1,e1,e2}
  int np = n_in - 5; if (np < 1) np = 1;
  auto Lp = [&](int idx) -> const float* {
    int j = 1 + ((idx - 1) % np);
    return (const float*)d_in[j];
  };
  auto Bp = [&](int leaf, int b) -> const float* {  // stacked [5,...] block leaf
    int j = 1 + ((leaf - 1) % np);
    size_t per = (size_t)in_sizes[j] / 5;
    return (const float*)d_in[j] + (size_t)b * per;
  };
  const int BA = 2, BF1S = 3, BF1T = 4, BF1V = 5, BF2S = 6, BF2T = 7, BF2V = 8, BFG = 9,
            BL0 = 10, BL1 = 11, BL2 = 12, BN1S = 13, BN1T = 14, BN1V = 15,
            BN2S = 16, BN2T = 17, BN2V = 18, BP0 = 19, BP1 = 20, BP2 = 21,
            BR1 = 22, BR2 = 23, BR3 = 24, BWD0 = 25, BWD1 = 26, BWD2 = 27,
            BWS0 = 28, BWS1 = 29, BWS2 = 30;
  const float* atom_emb = Lp(1);
  const float* deg_p0 = Lp(31); const float* deg_p1 = Lp(32); const float* deg_p2 = Lp(33);
  const float* deg_r1 = Lp(34); const float* deg_r2 = Lp(35); const float* deg_r3 = Lp(36);
  const float* fns = Lp(37);
  const float* head_e1 = Lp(41); const float* head_e2 = Lp(42);

  // workspace bump allocator
  float* wsp = (float*)d_ws;
  size_t off = 0;
  auto alloc = [&](size_t nelem) -> float* {
    float* p = wsp + off;
    off += (nelem + 63) & ~((size_t)63);
    return p;
  };
  float* sh   = alloc((size_t)E * 9);
  float* rbf  = alloc((size_t)E * 32);
  float* dist = alloc(E);
  float* uvec = alloc((size_t)E * 3);
  float* rh1  = alloc((size_t)E * 64);
  float* rh2  = alloc((size_t)E * 64);
  float* rw   = alloc((size_t)E * 960);
  float* s    = alloc((size_t)N * 128);
  float* v    = alloc((size_t)N * 192);
  float* t    = alloc((size_t)N * 160);
  float* ns   = alloc((size_t)N * 128);
  float* nv   = alloc((size_t)N * 192);
  float* nt   = alloc((size_t)N * 160);
  float* fs0  = alloc((size_t)N * 128);
  float* fd0  = alloc((size_t)N * 128);
  float* fs1  = alloc((size_t)N * 192);
  float* fd1  = alloc((size_t)N * 192);
  float* fs2  = alloc((size_t)N * 160);
  float* fd2  = alloc((size_t)N * 160);
  float* gate = alloc((size_t)N * 96);
  float* tmp0 = alloc((size_t)N * 128);
  float* msg0 = alloc((size_t)E * 128);
  float* msg1 = alloc((size_t)E * 192);
  float* msg2 = alloc((size_t)E * 160);
  float* d0b  = alloc((size_t)E * 224);
  float* d1b  = alloc((size_t)E * 1152);
  float* d2b  = alloc((size_t)E * 1760);
  float* h0   = alloc((size_t)E * 128);
  float* h1   = alloc((size_t)E * 192);
  float* h2   = alloc((size_t)E * 160);
  float* logit = alloc((size_t)E * 4);
  float* ex    = alloc((size_t)E * 4);
  float* alp   = alloc((size_t)E * 4);
  float* mxenc = alloc((size_t)N * 4);
  float* den   = alloc((size_t)N * 4);
  float* o0   = alloc((size_t)N * 128);
  float* o1   = alloc((size_t)N * 192);
  float* o2   = alloc((size_t)N * 160);
  float* hid  = alloc((size_t)N * 128);
  (void)ws_size;

  zero(stream, (float*)d_out, out_size);

  // ---- geometry + RBF ----
  edge_geom_kernel<<<(E + 255) / 256, 256, 0, stream>>>(pos, src, dst, sh, rbf, dist, uvec, E);

  // ---- degree embedding ----
  gemm(stream, rbf, deg_r1, rh1, E, 64, 32, 32, 1, 64, 1, 64, 1, 1, 0);
  gemm(stream, rh1, deg_r2, rh2, E, 64, 64, 64, 1, 64, 1, 64, 1, 1, 0);
  gemm(stream, rh2, deg_r3, rw, E, 224, 64, 64, 1, 224, 1, 960, 1, 0, 0);
  embed_kernel<<<(N * 128 + 255) / 256, 256, 0, stream>>>(atom_emb, z, s, N);
  zero(stream, v, (long)N * 192); zero(stream, t, (long)N * 160);
  zero(stream, tmp0, (long)N * 128); zero(stream, o1, (long)N * 192); zero(stream, o2, (long)N * 160);
  float invdeg = 1.f / sqrtf(15.57930851f);
  deg_scatter_kernel<<<E, 512, 0, stream>>>(rw, sh, dst, tmp0, o1, o2, E, invdeg);
  gemm(stream, tmp0, deg_p0, s, N, 128, 128, 128, 1, 128, 1, 128, 1, 0, 1);
  for (int m = 0; m < 3; ++m)
    gemm(stream, o1 + m, deg_p1, v + m, N, 64, 64, 192, 3, 64, 1, 192, 3, 0, 1);
  for (int m = 0; m < 5; ++m)
    gemm(stream, o2 + m, deg_p2, t + m, N, 32, 32, 160, 5, 32, 1, 160, 5, 0, 1);

  // ---- transformer blocks (5 stacked + final reusing block 4 weights) ----
  for (int blk = 0; blk < 6; ++blk) {
    int bb = blk < 5 ? blk : 4;
    // norm 1
    eqnorm_s_kernel<<<N, 128, 0, stream>>>(s, ns, Bp(BN1S, bb), N);
    eqnorm_vt_kernel<<<N, 256, 0, stream>>>(v, nv, Bp(BN1V, bb), N, 64, 3);
    eqnorm_vt_kernel<<<N, 256, 0, stream>>>(t, nt, Bp(BN1T, bb), N, 32, 5);
    // src/dst projections
    gemm(stream, ns, Bp(BWS0, bb), fs0, N, 128, 128, 128, 1, 128, 1, 128, 1, 0, 0);
    gemm(stream, ns, Bp(BWD0, bb), fd0, N, 128, 128, 128, 1, 128, 1, 128, 1, 0, 0);
    for (int m = 0; m < 3; ++m) {
      gemm(stream, nv + m, Bp(BWS1, bb), fs1 + m, N, 64, 64, 192, 3, 64, 1, 192, 3, 0, 0);
      gemm(stream, nv + m, Bp(BWD1, bb), fd1 + m, N, 64, 64, 192, 3, 64, 1, 192, 3, 0, 0);
    }
    for (int m = 0; m < 5; ++m) {
      gemm(stream, nt + m, Bp(BWS2, bb), fs2 + m, N, 32, 32, 160, 5, 32, 1, 160, 5, 0, 0);
      gemm(stream, nt + m, Bp(BWD2, bb), fd2 + m, N, 32, 32, 160, 5, 32, 1, 160, 5, 0, 0);
    }
    gather_msgs_kernel<<<E, 512, 0, stream>>>(fs0, fd0, fs1, fd1, fs2, fd2, src, dst,
                                              msg0, msg1, msg2, E);
    // radial weights
    gemm(stream, rbf, Bp(BR1, bb), rh1, E, 64, 32, 32, 1, 64, 1, 64, 1, 1, 0);
    gemm(stream, rh1, Bp(BR2, bb), rh2, E, 64, 64, 64, 1, 64, 1, 64, 1, 1, 0);
    gemm(stream, rh2, Bp(BR3, bb), rw, E, 960, 64, 64, 1, 960, 1, 960, 1, 0, 0);
    // tensor product + head linears
    dtp_kernel<<<E, 128, 0, stream>>>(msg0, msg1, msg2, sh, rw, d0b, d1b, d2b, E);
    gemm(stream, d0b, Bp(BL0, bb), h0, E, 128, 224, 224, 1, 128, 1, 128, 1, 0, 0);
    for (int m = 0; m < 3; ++m)
      gemm(stream, d1b + m, Bp(BL1, bb), h1 + m, E, 64, 384, 1152, 3, 64, 1, 192, 3, 0, 0);
    for (int m = 0; m < 5; ++m)
      gemm(stream, d2b + m, Bp(BL2, bb), h2 + m, E, 32, 352, 1760, 5, 32, 1, 160, 5, 0, 0);
    // attention softmax over segments
    zero(stream, mxenc, (long)N * 4); zero(stream, den, (long)N * 4);
    zero(stream, o0, (long)N * 128); zero(stream, o1, (long)N * 192); zero(stream, o2, (long)N * 160);
    logits_kernel<<<(E * 4 + 255) / 256, 256, 0, stream>>>(h0, Bp(BA, bb), logit,
                                                           (unsigned*)mxenc, dst, E);
    expsum_kernel<<<(E * 4 + 255) / 256, 256, 0, stream>>>(logit, (unsigned*)mxenc, dst, ex, den, E);
    alpha_kernel<<<(E * 4 + 255) / 256, 256, 0, stream>>>(ex, den, dst, alp, E);
    attn_scatter_kernel<<<E, 512, 0, stream>>>(h0, h1, h2, alp, dst, o0, o1, o2, E);
    // output projection, residual accumulate
    gemm(stream, o0, Bp(BP0, bb), s, N, 128, 128, 128, 1, 128, 1, 128, 1, 0, 1);
    for (int m = 0; m < 3; ++m)
      gemm(stream, o1 + m, Bp(BP1, bb), v + m, N, 64, 64, 192, 3, 64, 1, 192, 3, 0, 1);
    for (int m = 0; m < 5; ++m)
      gemm(stream, o2 + m, Bp(BP2, bb), t + m, N, 32, 32, 160, 5, 32, 1, 160, 5, 0, 1);
    // FFN
    eqnorm_s_kernel<<<N, 128, 0, stream>>>(s, ns, Bp(BN2S, bb), N);
    eqnorm_vt_kernel<<<N, 256, 0, stream>>>(v, nv, Bp(BN2V, bb), N, 64, 3);
    eqnorm_vt_kernel<<<N, 256, 0, stream>>>(t, nt, Bp(BN2T, bb), N, 32, 5);
    gemm(stream, ns, Bp(BFG, bb), gate, N, 96, 128, 128, 1, 96, 1, 96, 1, 3, 0);
    gemm(stream, ns, Bp(BF1S, bb), tmp0, N, 128, 128, 128, 1, 128, 1, 128, 1, 1, 0);
    for (int m = 0; m < 3; ++m)
      gemm(stream, nv + m, Bp(BF1V, bb), fs1 + m, N, 64, 64, 192, 3, 64, 1, 192, 3, 0, 0);
    for (int m = 0; m < 5; ++m)
      gemm(stream, nt + m, Bp(BF1T, bb), fs2 + m, N, 32, 32, 160, 5, 32, 1, 160, 5, 0, 0);
    gate_apply_kernel<<<N, 384, 0, stream>>>(fs1, fs2, gate, N);
    gemm(stream, tmp0, Bp(BF2S, bb), s, N, 128, 128, 128, 1, 128, 1, 128, 1, 0, 1);
    for (int m = 0; m < 3; ++m)
      gemm(stream, fs1 + m, Bp(BF2V, bb), v + m, N, 64, 64, 192, 3, 64, 1, 192, 3, 0, 1);
    for (int m = 0; m < 5; ++m)
      gemm(stream, fs2 + m, Bp(BF2T, bb), t + m, N, 32, 32, 160, 5, 32, 1, 160, 5, 0, 1);
  }

  // ---- head: energy per graph ----
  eqnorm_s_kernel<<<N, 128, 0, stream>>>(s, ns, fns, N);
  gemm(stream, ns, head_e1, hid, N, 128, 128, 128, 1, 512, 1, 128, 1, 1, 0);
  node_energy_kernel<<<(N + 255) / 256, 256, 0, stream>>>(hid, head_e2, batch, (float*)d_out, N,
                                                          1.f / sqrtf(18.03065905f));

  // ---- forces (backward through cutoff/RBF chain) ----
  int foff = out_size - N * 3; if (foff < 0) foff = 0;
  force_kernel<<<(E + 255) / 256, 256, 0, stream>>>(dist, uvec, rw, src, dst,
                                                    (float*)d_out + foff, E);
}
